// TAILMIL_16131897163885
// MI455X (gfx1250) — matrix operand order repaired and checked by reading, not compiled, not run on hardware
//
#include <hip/hip_runtime.h>

typedef _Float16 h16;
typedef __attribute__((ext_vector_type(16))) _Float16 v16h;
typedef __attribute__((ext_vector_type(8)))  _Float16 v8h;
typedef __attribute__((ext_vector_type(8)))  float    v8f;

__device__ __forceinline__ float leakyf(float v) { return v > 0.f ? v : 0.01f * v; }
__device__ __forceinline__ float sigmf(float v)  { return 1.f / (1.f + __expf(-v)); }

// ---- WMMA fragment helpers (CDNA5 16x16x32 f16 layouts per CDNA5 ISA) ----
// A: row-major [M][lda] halves. Per lane: rows m = mBase+(lane&15); halves at
// kb+koff+{0..7} and kb+koff+{16..23}, koff = (lane&16)?8:0 -> two 16B vector loads.
__device__ __forceinline__ v16h load_a_frag(const h16* A, int lda, int mBase, int kb, int lane) {
  int m = mBase + (lane & 15);
  const h16* row = A + (size_t)m * lda + kb + ((lane & 16) ? 8 : 0);
  v8h lo = *(const v8h*)row;          // K = koff + 0..7
  v8h hi = *(const v8h*)(row + 16);   // K = koff + 16..23
  return __builtin_shufflevector(lo, hi, 0, 1, 2, 3, 4, 5, 6, 7,
                                 8, 9, 10, 11, 12, 13, 14, 15);
}
// B stored column-major: Bc[n][k] (ldk halves per column). Per lane: n = nBase+(lane&15),
// halves at kb+koff+{0..7} and kb+koff+{16..23}, koff = (lane&16)?8:0 -> two 16B vector loads.
__device__ __forceinline__ v16h load_b_fragT(const h16* Bc, int ldk, int nBase, int kb, int lane) {
  int n  = nBase + (lane & 15);
  const h16* p = Bc + (size_t)n * ldk + kb + ((lane & 16) ? 8 : 0);
  v8h lo = *(const v8h*)p;
  v8h hi = *(const v8h*)(p + 16);
  return __builtin_shufflevector(lo, hi, 0, 1, 2, 3, 4, 5, 6, 7,
                                 8, 9, 10, 11, 12, 13, 14, 15);
}
__device__ __forceinline__ v8f wmma16(v16h a, v16h b, v8f c) {
  return __builtin_amdgcn_wmma_f32_16x16x32_f16(false, a, false, b, (short)0, c, false, false);
}
#define V8F_ZERO {0.f,0.f,0.f,0.f,0.f,0.f,0.f,0.f}

// =====================  Stage 0: Wk2 -> f16 (already [n][k] layout)  =====================
__global__ void tailmil_cvt_wk2(const float* __restrict__ Wk2, h16* __restrict__ wk2h) {
  int idx = blockIdx.x * 256 + threadIdx.x;
  if (idx < 2432 * 64) wk2h[idx] = (h16)Wk2[idx];
}

// =====================  Stage A: per-(b,d) feature chain  =====================
__global__ __launch_bounds__(128)
void tailmil_stageA(const float* __restrict__ x,  const float* __restrict__ W1, const float* __restrict__ b1,
                    const float* __restrict__ W2, const float* __restrict__ b2,
                    const float* __restrict__ Wl, const float* __restrict__ bl,
                    const float* __restrict__ Wb1,const float* __restrict__ bb1,
                    const float* __restrict__ lng,const float* __restrict__ lnb,
                    const float* __restrict__ Wb2,const float* __restrict__ bb2,
                    const float* __restrict__ Wq, const float* __restrict__ bq,
                    const float* __restrict__ Wk, const float* __restrict__ bk,
                    const float* __restrict__ Wv1,const float* __restrict__ bv1,
                    const float* __restrict__ lnvg,const float* __restrict__ lnvb,
                    const float* __restrict__ Wv2,const float* __restrict__ bv2,
                    h16* __restrict__ e_h, h16* __restrict__ wk_ws,
                    float* __restrict__ wq_ws, float* __restrict__ wv_ws) {
  __shared__ float smem[16384];                 // 64 KB, 4 reusable regions
  const int tid  = threadIdx.x;
  const int lane = tid & 31;
  const int wid  = tid >> 5;                    // 4 waves
  const int bd   = blockIdx.x;                  // b*38 + d
  const int d    = bd % 38;

  float* R0 = smem;                             // f1 -> e(f32) -> acml
  h16*   Ah = (h16*)(smem + 4096);              // feat / A fragments (f16)
  h16*   Bh = (h16*)(smem + 8192);              // B operand, column-major [n][k] (f16)
  float* R3 = smem + 12288;                     // x -> vpre -> h -> g

  // Phase 1: load x window
  for (int i = tid; i < 640; i += 128) R3[i] = x[(size_t)bd * 640 + i];
  __syncthreads();

  // Phase 2: f1[s][o] = leaky(x_window . W1)
  for (int i = tid; i < 4096; i += 128) {
    int s = i >> 6, o = i & 63;
    float acc = b1[d * 64 + o];
    const float* w  = W1 + (size_t)(d * 64 + o) * 10;
    const float* xr = R3 + s * 10;
#pragma unroll
    for (int wi = 0; wi < 10; ++wi) acc += xr[wi] * w[wi];
    R0[i] = leakyf(acc);
  }
  __syncthreads();

  // Phase 3: conv(k=3,pad=1,4ch) + leaky + maxpool(3,2): f1 -> feat[64][128] (f16, pad 124->128)
  {
    float w2c[4][3], b2c[4];
#pragma unroll
    for (int c = 0; c < 4; ++c) {
      b2c[c] = b2[d * 4 + c];
#pragma unroll
      for (int r = 0; r < 3; ++r) w2c[c][r] = W2[(d * 4 + c) * 3 + r];
    }
    for (int i = tid; i < 64 * 124; i += 128) {
      int s = i / 124, f = i % 124, c = f / 31, p = f % 31;
      float best = -3.4e38f;
#pragma unroll
      for (int dt = 0; dt < 3; ++dt) {
        int t = 2 * p + dt;
        float cv = b2c[c];
#pragma unroll
        for (int r = 0; r < 3; ++r) {
          int ix = t - 1 + r;
          if (ix >= 0 && ix < 64) cv += R0[s * 64 + ix] * w2c[c][r];
        }
        best = fmaxf(best, leakyf(cv));
      }
      Ah[s * 128 + f] = (h16)best;
    }
    for (int i = tid; i < 64 * 4; i += 128) Ah[(i >> 2) * 128 + 124 + (i & 3)] = (h16)0.f;
    // B = Wl[d] column-major [n=64][k=128] : contiguous copy of Wl rows, pad k>=124
    for (int i = tid; i < 64 * 128; i += 128) {
      int n = i >> 7, k = i & 127;
      Bh[i] = (k < 124) ? (h16)Wl[((size_t)d * 64 + n) * 124 + k] : (h16)0.f;
    }
  }
  __syncthreads();

  // Phase 4: e = leaky(feat @ WlT + bl)  [64x64, K=128]  (WMMA)
  {
    int mBase = wid * 16;
    for (int nt = 0; nt < 4; ++nt) {
      v8f acc = V8F_ZERO;
      for (int kt = 0; kt < 4; ++kt)
        acc = wmma16(load_a_frag(Ah, 128, mBase, kt * 32, lane),
                     load_b_fragT(Bh, 128, nt * 16, kt * 32, lane), acc);
      int n = nt * 16 + (lane & 15), mo = (lane & 16) ? 8 : 0;
#pragma unroll
      for (int r = 0; r < 8; ++r)
        R0[(mBase + mo + r) * 64 + n] = leakyf(acc[r] + bl[d * 64 + n]);
    }
  }
  __syncthreads();

  // Phase 5: write e (f16) to workspace; wq = sigmoid(e . Wq + bq)
  for (int i = tid; i < 4096; i += 128) e_h[(size_t)bd * 4096 + i] = (h16)R0[i];
  if (tid < 64) {
    float acc = bq[0];
    for (int o = 0; o < 64; ++o) acc += R0[tid * 64 + o] * Wq[o];
    wq_ws[(size_t)bd * 64 + tid] = sigmf(acc);
  }
  __syncthreads();

  // Phase 6: Ah <- e (f16) [64][64]; Bh <- Wv1[d] column-major [n=32][k=64] (direct layout)
  for (int i = tid; i < 4096; i += 128) Ah[i] = (h16)R0[i];
  for (int i = tid; i < 2048; i += 128) Bh[i] = (h16)Wv1[(size_t)d * 2048 + i];
  __syncthreads();

  // Phase 7: vpre = e @ Wv1T + bv1  [64x32, K=64]  (WMMA)
  {
    int mBase = wid * 16;
    for (int nt = 0; nt < 2; ++nt) {
      v8f acc = V8F_ZERO;
      for (int kt = 0; kt < 2; ++kt)
        acc = wmma16(load_a_frag(Ah, 64, mBase, kt * 32, lane),
                     load_b_fragT(Bh, 64, nt * 16, kt * 32, lane), acc);
      int n = nt * 16 + (lane & 15), mo = (lane & 16) ? 8 : 0;
#pragma unroll
      for (int r = 0; r < 8; ++r)
        R3[(mBase + mo + r) * 32 + n] = acc[r] + bv1[d * 32 + n];
    }
  }
  __syncthreads();

  // Phase 8: v = leaky(LN32(vpre)); wv = relu(v . Wv2 + bv2). Also Bh <- Wb1 (direct copy, [p][o])
  if (tid < 64) {
    int s = tid;
    float mu = 0.f;
    for (int h = 0; h < 32; ++h) mu += R3[s * 32 + h];
    mu *= (1.f / 32.f);
    float m2 = 0.f;
    for (int h = 0; h < 32; ++h) { float t = R3[s * 32 + h] - mu; m2 += t * t; }
    float inv = rsqrtf(m2 * (1.f / 32.f) + 1e-5f);
    float acc = bv2[d];
    for (int h = 0; h < 32; ++h) {
      float vv = leakyf(lnvg[d * 32 + h] * (R3[s * 32 + h] - mu) * inv + lnvb[d * 32 + h]);
      acc += vv * Wv2[d * 32 + h];
    }
    wv_ws[(size_t)bd * 64 + s] = fmaxf(acc, 0.f);
  }
  for (int i = tid; i < 4096; i += 128) Bh[i] = (h16)Wb1[i];
  __syncthreads();

  // Phase 9: hpre = e @ Wb1T + bb1  [64x64, K=64]  (WMMA)
  {
    int mBase = wid * 16;
    for (int nt = 0; nt < 4; ++nt) {
      v8f acc = V8F_ZERO;
      for (int kt = 0; kt < 2; ++kt)
        acc = wmma16(load_a_frag(Ah, 64, mBase, kt * 32, lane),
                     load_b_fragT(Bh, 64, nt * 16, kt * 32, lane), acc);
      int n = nt * 16 + (lane & 15), mo = (lane & 16) ? 8 : 0;
#pragma unroll
      for (int r = 0; r < 8; ++r)
        R3[(mBase + mo + r) * 64 + n] = acc[r] + bb1[n];
    }
  }
  __syncthreads();

  // Phase 10: h = leaky(LN64(hpre)) in place
  if (tid < 64) {
    int s = tid;
    float mu = 0.f;
    for (int p = 0; p < 64; ++p) mu += R3[s * 64 + p];
    mu *= (1.f / 64.f);
    float m2 = 0.f;
    for (int p = 0; p < 64; ++p) { float t = R3[s * 64 + p] - mu; m2 += t * t; }
    float inv = rsqrtf(m2 * (1.f / 64.f) + 1e-5f);
    for (int p = 0; p < 64; ++p)
      R3[s * 64 + p] = leakyf(lng[p] * (R3[s * 64 + p] - mu) * inv + lnb[p]);
  }
  __syncthreads();

  // Phase 11: Ah <- h (f16); Bh <- Wb2 (direct copy, [q][p])
  for (int i = tid; i < 4096; i += 128) Ah[i] = (h16)R3[i];
  for (int i = tid; i < 4096; i += 128) Bh[i] = (h16)Wb2[i];
  __syncthreads();

  // Phase 12: g = leaky(h @ Wb2T + bb2)  (WMMA) -> R3
  {
    int mBase = wid * 16;
    for (int nt = 0; nt < 4; ++nt) {
      v8f acc = V8F_ZERO;
      for (int kt = 0; kt < 2; ++kt)
        acc = wmma16(load_a_frag(Ah, 64, mBase, kt * 32, lane),
                     load_b_fragT(Bh, 64, nt * 16, kt * 32, lane), acc);
      int n = nt * 16 + (lane & 15), mo = (lane & 16) ? 8 : 0;
#pragma unroll
      for (int r = 0; r < 8; ++r)
        R3[(mBase + mo + r) * 64 + n] = leakyf(acc[r] + bb2[n]);
    }
  }
  __syncthreads();

  // Phase 13: acml = cumsum_s(g) -> R0
  if (tid < 64) {
    float run = 0.f;
    for (int s = 0; s < 64; ++s) { run += R3[s * 64 + tid]; R0[s * 64 + tid] = run; }
  }
  __syncthreads();

  // Phase 14: Ah <- acml (f16); Bh <- Wk column-major [n=48][k=64] (rows 38..47 zero)
  for (int i = tid; i < 4096; i += 128) Ah[i] = (h16)R0[i];
  for (int i = tid; i < 48 * 64; i += 128) {
    int n = i >> 6;
    Bh[i] = (n < 38) ? (h16)Wk[i] : (h16)0.f;
  }
  __syncthreads();

  // Phase 15: wk = sigmoid(acml @ WkT + bk)  [64x38, K=64]  (WMMA), store f16
  {
    int mBase = wid * 16;
    for (int nt = 0; nt < 3; ++nt) {
      v8f acc = V8F_ZERO;
      for (int kt = 0; kt < 2; ++kt)
        acc = wmma16(load_a_frag(Ah, 64, mBase, kt * 32, lane),
                     load_b_fragT(Bh, 64, nt * 16, kt * 32, lane), acc);
      int n = nt * 16 + (lane & 15), mo = (lane & 16) ? 8 : 0;
      if (n < 38) {
#pragma unroll
        for (int r = 0; r < 8; ++r) {
          int s = mBase + mo + r;
          wk_ws[((size_t)bd * 64 + s) * 38 + n] = (h16)sigmf(acc[r] + bk[n]);
        }
      }
    }
  }
}

// =====================  Stage B: per-b attention + bag GEMM  =====================
__global__ __launch_bounds__(128)
void tailmil_stageB(const h16* __restrict__ e_h, const h16* __restrict__ wk_ws,
                    const float* __restrict__ wq_ws, const float* __restrict__ wv_ws,
                    const h16* __restrict__ wk2h, const float* __restrict__ Wq2,
                    const float* __restrict__ bq2, const float* __restrict__ bk2,
                    float* __restrict__ out) {
  __shared__ float smem[8192];                  // 32 KB
  const int tid = threadIdx.x, lane = tid & 31, wid = tid >> 5;
  const int b = blockIdx.x;
  float* wqT  = smem;                           // [k=38][s=64]
  float* L    = smem + 2432;                    // [s][j] stride 40
  float* out1 = smem + 8000;                    // [64]

  // B1: load wq for this b
  for (int i = tid; i < 2432; i += 128) wqT[i] = wq_ws[(size_t)b * 2432 + i];
  __syncthreads();

  // B2: att1 logits[s][j] = sum_k wq[k][s] * wk[b,j,s,k] / sqrt(38)
  const float scale1 = rsqrtf(38.f);
  for (int i = tid; i < 64 * 38; i += 128) {
    int s = i / 38, j = i % 38;
    const h16* row = wk_ws + ((size_t)(b * 38 + j) * 64 + s) * 38;
    float acc = 0.f;
    for (int k = 0; k < 38; ++k) acc += wqT[k * 64 + s] * (float)row[k];
    L[s * 40 + j] = acc * scale1;
  }
  __syncthreads();

  // B3: softmax over j, contributes1, output1
  if (tid < 64) {
    int s = tid;
    float mx = -3.4e38f;
    for (int j = 0; j < 38; ++j) mx = fmaxf(mx, L[s * 40 + j]);
    float sum = 0.f;
    for (int j = 0; j < 38; ++j) sum += __expf(L[s * 40 + j] - mx);
    float inv = 1.f / sum;
    float o1 = 0.f;
    for (int j = 0; j < 38; ++j) {
      float c1 = __expf(L[s * 40 + j] - mx) * inv * wv_ws[((size_t)(b * 38 + j)) * 64 + s];
      out[16640 + ((size_t)b * 64 + s) * 38 + j] = c1;   // contributes1
      o1 += c1;
    }
    out1[s] = o1;
  }
  __syncthreads();

  // B4: wq_a[s] = sigmoid(ef[s] . Wq2 + bq2), 2 threads per s
  float* wqa  = smem + 4096;
  float* part = smem + 4224;
  {
    int s = tid >> 1, hsel = tid & 1;
    const h16* eb = e_h + (size_t)b * 38 * 4096;
    float acc = 0.f;
    int f0 = hsel * 1216;
    for (int f = f0; f < f0 + 1216; ++f) {
      int dd = f >> 6, o = f & 63;
      acc += (float)eb[(size_t)dd * 4096 + s * 64 + o] * Wq2[f];
    }
    part[tid] = acc;
  }
  __syncthreads();
  if (tid < 64) wqa[tid] = sigmf(part[2 * tid] + part[2 * tid + 1] + bq2[0]);
  __syncthreads();

  // B5: Mm[t][s] = ef[t] . Wk2[s]  [64x64, K=2432]  (WMMA; A from global f16 e, B = wk2h [n][k])
  float* Mm = smem;                             // overwrite wqT/L (dead)
  {
    int mBase = wid * 16;
    const h16* eb = e_h + (size_t)b * 38 * 4096;
    for (int nt = 0; nt < 4; ++nt) {
      v8f acc = V8F_ZERO;
      for (int kt = 0; kt < 76; ++kt) {
        int dd = kt >> 1, ob = (kt & 1) * 32;   // each 32-wide k-tile sits in one channel d
        v16h a  = load_a_frag(eb + (size_t)dd * 4096 + ob, 64, mBase, 0, lane);
        v16h bf = load_b_fragT(wk2h, 2432, nt * 16, kt * 32, lane);
        acc = wmma16(a, bf, acc);
      }
      int n = nt * 16 + (lane & 15), mo = (lane & 16) ? 8 : 0;
#pragma unroll
      for (int r = 0; r < 8; ++r) Mm[(mBase + mo + r) * 64 + n] = acc[r];
    }
  }
  __syncthreads();

  // B6: att2 logits, softmax over t, contributes2, output2
  float* l2  = smem + 4160;
  float* l2b = smem + 4288;
  const float scale2 = rsqrtf(640.f);
  if (tid < 64) {
    int t = tid;
    float acc = 0.f;
    for (int s = 0; s < 64; ++s) acc += wqa[s] * sigmf(Mm[t * 64 + s] + bk2[s]);
    l2[t] = acc * scale2;
  }
  __syncthreads();
  if (tid < 64) {
    float mx = -3.4e38f;
    for (int t2 = 0; t2 < 64; ++t2) mx = fmaxf(mx, l2[t2]);
    l2b[tid] = __expf(l2[tid] - mx);
  }
  __syncthreads();
  if (tid < 64) {
    float sum = 0.f;
    for (int t2 = 0; t2 < 64; ++t2) sum += l2b[t2];
    float c2 = out1[tid] * (l2b[tid] / sum);
    out[256 + (size_t)b * 64 + tid] = c2;       // contributes2
    l2[tid] = c2;
  }
  __syncthreads();
  if (tid == 0) {
    float s2 = 0.f;
    for (int t2 = 0; t2 < 64; ++t2) s2 += l2[t2];
    out[b] = s2;                                // output2
  }
}

// =====================  Launch  =====================
extern "C" void kernel_launch(void* const* d_in, const int* in_sizes, int n_in,
                              void* d_out, int out_size, void* d_ws, size_t ws_size,
                              hipStream_t stream) {
  const float* x    = (const float*)d_in[0];
  const float* W1   = (const float*)d_in[1];
  const float* b1   = (const float*)d_in[2];
  const float* W2   = (const float*)d_in[3];
  const float* b2   = (const float*)d_in[4];
  const float* Wl   = (const float*)d_in[5];
  const float* bl   = (const float*)d_in[6];
  const float* Wb1  = (const float*)d_in[7];
  const float* bb1  = (const float*)d_in[8];
  const float* lng  = (const float*)d_in[9];
  const float* lnb  = (const float*)d_in[10];
  const float* Wb2  = (const float*)d_in[11];
  const float* bb2  = (const float*)d_in[12];
  const float* Wq   = (const float*)d_in[13];
  const float* bq   = (const float*)d_in[14];
  const float* Wk   = (const float*)d_in[15];
  const float* bk   = (const float*)d_in[16];
  const float* Wv1  = (const float*)d_in[17];
  const float* bv1  = (const float*)d_in[18];
  const float* lnvg = (const float*)d_in[19];
  const float* lnvb = (const float*)d_in[20];
  const float* Wv2  = (const float*)d_in[21];
  const float* bv2  = (const float*)d_in[22];
  const float* Wq2  = (const float*)d_in[23];
  const float* bq2  = (const float*)d_in[24];
  const float* Wk2  = (const float*)d_in[25];
  const float* bk2  = (const float*)d_in[26];

  char* ws = (char*)d_ws;
  const size_t off_eh   = 0;                         // 256*38*64*64 f16 = 79,691,776 B
  const size_t off_wk   = off_eh + 79691776ull;      // 256*38*64*38 f16 = 47,316,992 B
  const size_t off_wq   = off_wk + 47316992ull;      // 256*38*64 f32 = 2,490,368 B
  const size_t off_wv   = off_wq + 2490368ull;
  const size_t off_wk2h = off_wv + 2490368ull;       // 2432*64 f16

  h16*   e_h   = (h16*)(ws + off_eh);
  h16*   wk_ws = (h16*)(ws + off_wk);
  float* wq_ws = (float*)(ws + off_wq);
  float* wv_ws = (float*)(ws + off_wv);
  h16*   wk2h  = (h16*)(ws + off_wk2h);

  tailmil_cvt_wk2<<<(2432 * 64 + 255) / 256, 256, 0, stream>>>(Wk2, wk2h);
  tailmil_stageA<<<256 * 38, 128, 0, stream>>>(x, W1, b1, W2, b2, Wl, bl, Wb1, bb1, lng, lnb,
                                               Wb2, bb2, Wq, bq, Wk, bk, Wv1, bv1, lnvg, lnvb,
                                               Wv2, bv2, e_h, wk_ws, wq_ws, wv_ws);
  tailmil_stageB<<<256, 128, 0, stream>>>(e_h, wk_ws, wq_ws, wv_ws, wk2h, Wq2, bq2, bk2,
                                          (float*)d_out);
}
